// PointNetPolylineEncoder_1606317769339
// MI455X (gfx1250) — matrix operand order repaired and checked by reading, not compiled, not run on hardware
//
#include <hip/hip_runtime.h>

// ---- CDNA5 WMMA types (bf16 16x16x32, f32 accum) ----
typedef __bf16 v16bf __attribute__((ext_vector_type(16)));
typedef float  v8f   __attribute__((ext_vector_type(8)));
typedef unsigned short v16u __attribute__((ext_vector_type(16)));

#define BN_EPS 1e-5f

__device__ __forceinline__ unsigned short f2bf(float f) {
  __bf16 h = (__bf16)f;                       // native v_cvt, RNE
  return __builtin_bit_cast(unsigned short, h);
}
__device__ __forceinline__ float bf2f(unsigned short h) {
  return __uint_as_float(((unsigned int)h) << 16);  // exact widening
}

#define MMA(a, b, c) \
  __builtin_amdgcn_wmma_f32_16x16x32_bf16(false, (a), false, (b), (short)0, (c), false, false)
#define BC(x) __builtin_bit_cast(v16bf, (x))

// ---- LDS layout (units: u16) ----
// B-fragment regions are 512 u16 (32 lanes x 16 elems = 1KB) each.
#define WPRE_BASE   0          //  4 regions (nt 0..3), K padded 9->32
#define W1F_BASE    2048       // 16 regions (kc 0..3, nt 0..3)
#define W2F_BASE    10240      //  8 regions (kc 0..1, nt 0..3)
#define WAVE_BASE   14336      // per-wave: 4 A-fragment regions (2048) + 64 pooled
#define WAVE_STRIDE 2112
#define LDS_U16     (14336 + 8 * 2112)   // 31232 u16 = 62464 B

__global__ __launch_bounds__(256)
void pointnet_fused(const float* __restrict__ poly,      // [16384,32,9]
                    const float* __restrict__ Wpre,      // [9,64]
                    const float* __restrict__ G0, const float* __restrict__ B0,
                    const float* __restrict__ M0, const float* __restrict__ V0,
                    const float* __restrict__ W1,        // [128,64]
                    const float* __restrict__ G1, const float* __restrict__ B1,
                    const float* __restrict__ M1, const float* __restrict__ V1,
                    const float* __restrict__ W2,        // [64,64]
                    const float* __restrict__ G2, const float* __restrict__ B2,
                    const float* __restrict__ M2, const float* __restrict__ V2,
                    const float* __restrict__ W3, const float* __restrict__ b3,  // [64,64],[64]
                    const float* __restrict__ W4, const float* __restrict__ b4,  // [64,128],[128]
                    const unsigned char* __restrict__ maskp,  // [16384,32] bool
                    float* __restrict__ out)             // [16384,128]
{
  __shared__ __align__(32) unsigned short lds[LDS_U16];
  const int tid = threadIdx.x;

  // ---------------- cooperative weight prep: f32 -> bf16 B-fragments ----------------
  // B-fragment element (lane l, elem i) holds W[kc*32 + (i>=8?16:0) + (l>=16?8:0) + (i&7)][nt*16 + (l&15)]
  for (int idx = tid; idx < 2048; idx += 256) {           // W_pre (K padded to 32)
    int nt = idx >> 9, l = (idx >> 4) & 31, i = idx & 15;
    int K = ((i & 8) << 1) + ((l & 16) >> 1) + (i & 7);
    int ch = nt * 16 + (l & 15);
    lds[WPRE_BASE + idx] = (K < 9) ? f2bf(Wpre[K * 64 + ch]) : (unsigned short)0;
  }
  for (int idx = tid; idx < 8192; idx += 256) {           // W1 (K=128)
    int reg = idx >> 9, l = (idx >> 4) & 31, i = idx & 15;
    int kc = reg >> 2, nt = reg & 3;
    int K = kc * 32 + ((i & 8) << 1) + ((l & 16) >> 1) + (i & 7);
    int ch = nt * 16 + (l & 15);
    lds[W1F_BASE + idx] = f2bf(W1[K * 64 + ch]);
  }
  for (int idx = tid; idx < 4096; idx += 256) {           // W2 (K=64)
    int reg = idx >> 9, l = (idx >> 4) & 31, i = idx & 15;
    int kc = reg >> 2, nt = reg & 3;
    int K = kc * 32 + ((i & 8) << 1) + ((l & 16) >> 1) + (i & 7);
    int ch = nt * 16 + (l & 15);
    lds[W2F_BASE + idx] = f2bf(W2[K * 64 + ch]);
  }
  __syncthreads();

  // ---------------- per-wave: one polyline ----------------
  const int lane = tid & 31;
  const int wave = tid >> 5;
  const int h    = lane >> 4;      // A/B half selector
  const int l15  = lane & 15;
  const int p    = blockIdx.x * 8 + wave;      // 0..16383
  const int AW   = WAVE_BASE + wave * WAVE_STRIDE;
  const int POOL = AW + 2048;

  auto ldfrag = [&](int base) -> v16bf {
    return __builtin_bit_cast(v16bf,
        *reinterpret_cast<const v16u*>(&lds[base + lane * 16]));
  };

  // ---- async-stage this wave's raw points into its (idle) A-region: 288 f32, coalesced
  {
    unsigned lbase = (unsigned)(unsigned long long)&lds[AW];   // LDS byte address
    const float* gbase = poly + (size_t)p * 288;
    #pragma unroll
    for (int j = 0; j < 9; ++j) {
      asm volatile("global_load_async_to_lds_b32 %0, %1, off"
                   :: "v"(lbase + (unsigned)(lane * 4 + j * 128)),
                      "v"(gbase + lane + j * 32)
                   : "memory");
    }
  }

  // point-validity bitmask for this polyline (overlaps with async copy)
  unsigned int mball =
      (unsigned int)__ballot((int)(maskp[(size_t)p * 32 + lane] != 0));

  asm volatile("s_wait_asynccnt 0" ::: "memory");
  const float* stage = (const float*)&lds[AW];   // staged points, row-major [32][9]

  // ---- stage 1 A fragments from staged LDS (rows = points, K = channel, pad 9->32)
  v16u a1u[2];
  #pragma unroll
  for (int mt = 0; mt < 2; ++mt) {
    const float* row = stage + (mt * 16 + l15) * 9;
    v16u f;
    #pragma unroll
    for (int i = 0; i < 16; ++i) {
      int K = (i < 8) ? (h * 8 + i) : (16 + h * 8 + (i - 8));  // A 16x32 bf16 layout
      f[i] = (K < 9) ? f2bf(row[K]) : (unsigned short)0;
    }
    a1u[mt] = f;
  }

  const v8f vzero = {};

  // ---- stage 1 GEMM: [32x32pad] @ [32x64] ----
  v8f acc1[2][4];
  #pragma unroll
  for (int mt = 0; mt < 2; ++mt)
    #pragma unroll
    for (int nt = 0; nt < 4; ++nt)
      acc1[mt][nt] = MMA(BC(a1u[mt]), ldfrag(WPRE_BASE + nt * 512), vzero);

  // ---- stage 1 post: BN + ReLU + mask, scatter into next-stage A fragments, pool max
  {
    #pragma unroll
    for (int nt = 0; nt < 4; ++nt) {
      int c = nt * 16 + l15;                       // output feature 0..63
      float sc = G0[c] * rsqrtf(V0[c] + BN_EPS);
      float tc = B0[c] - M0[c] * sc;
      int kc  = c >> 5;                            // next-stage K chunk
      int K32 = c & 31;
      int hb  = (K32 >> 3) & 1;                    // A-layout lane half for this K
      int ie  = (K32 & 7) + ((K32 & 16) >> 1);     // A-layout element index
      float pmv = 0.f;
      #pragma unroll
      for (int mt = 0; mt < 2; ++mt) {
        int rb = AW + (kc * 2 + mt) * 512;
        #pragma unroll
        for (int r = 0; r < 8; ++r) {
          int Mrow = mt * 16 + h * 8 + r;          // C/D layout row
          float y = acc1[mt][nt][r] * sc + tc;
          y = fmaxf(y, 0.f);
          y = ((mball >> Mrow) & 1u) ? y : 0.f;
          pmv = fmaxf(pmv, y);
          lds[rb + (h * 8 + r + (hb << 4)) * 16 + ie] = f2bf(y);
        }
      }
      pmv = fmaxf(pmv, __shfl_xor(pmv, 16));
      if (lane < 16) lds[POOL + nt * 16 + l15] = f2bf(pmv);
    }
  }

  // ---- stage 2 A operands: x-part from swizzled LDS, pooled-part built in regs
  v16u a2x[2][2];
  #pragma unroll
  for (int kc = 0; kc < 2; ++kc)
    #pragma unroll
    for (int mt = 0; mt < 2; ++mt)
      a2x[kc][mt] = *reinterpret_cast<const v16u*>(
          &lds[AW + (kc * 2 + mt) * 512 + lane * 16]);
  v16u a2p[2];                                      // pooled broadcast rows (all M equal)
  #pragma unroll
  for (int c = 0; c < 2; ++c) {
    v16u f;
    #pragma unroll
    for (int i = 0; i < 16; ++i) {
      int K = (i < 8) ? (h * 8 + i) : (16 + h * 8 + (i - 8));
      f[i] = lds[POOL + c * 32 + K];
    }
    a2p[c] = f;
  }

  // ---- stage 2 GEMM: [32x128] @ [128x64] ----
  v8f acc2[2][4];
  #pragma unroll
  for (int mt = 0; mt < 2; ++mt)
    #pragma unroll
    for (int nt = 0; nt < 4; ++nt) {
      v8f c = MMA(BC(a2x[0][mt]), ldfrag(W1F_BASE + (0 * 4 + nt) * 512), vzero);
      c = MMA(BC(a2x[1][mt]), ldfrag(W1F_BASE + (1 * 4 + nt) * 512), c);
      c = MMA(BC(a2p[0]),     ldfrag(W1F_BASE + (2 * 4 + nt) * 512), c);
      c = MMA(BC(a2p[1]),     ldfrag(W1F_BASE + (3 * 4 + nt) * 512), c);
      acc2[mt][nt] = c;
    }

  // ---- stage 2 post: BN + ReLU + mask -> A fragments (reuse same regions)
  {
    #pragma unroll
    for (int nt = 0; nt < 4; ++nt) {
      int c = nt * 16 + l15;
      float sc = G1[c] * rsqrtf(V1[c] + BN_EPS);
      float tc = B1[c] - M1[c] * sc;
      int kc = c >> 5, K32 = c & 31;
      int hb = (K32 >> 3) & 1;
      int ie = (K32 & 7) + ((K32 & 16) >> 1);
      #pragma unroll
      for (int mt = 0; mt < 2; ++mt) {
        int rb = AW + (kc * 2 + mt) * 512;
        #pragma unroll
        for (int r = 0; r < 8; ++r) {
          int Mrow = mt * 16 + h * 8 + r;
          float y = acc2[mt][nt][r] * sc + tc;
          y = fmaxf(y, 0.f);
          y = ((mball >> Mrow) & 1u) ? y : 0.f;
          lds[rb + (h * 8 + r + (hb << 4)) * 16 + ie] = f2bf(y);
        }
      }
    }
  }

  // ---- stage 3 GEMM: [32x64] @ [64x64] ----
  v16u a3[2][2];
  #pragma unroll
  for (int kc = 0; kc < 2; ++kc)
    #pragma unroll
    for (int mt = 0; mt < 2; ++mt)
      a3[kc][mt] = *reinterpret_cast<const v16u*>(
          &lds[AW + (kc * 2 + mt) * 512 + lane * 16]);
  v8f acc3[2][4];
  #pragma unroll
  for (int mt = 0; mt < 2; ++mt)
    #pragma unroll
    for (int nt = 0; nt < 4; ++nt) {
      v8f c = MMA(BC(a3[0][mt]), ldfrag(W2F_BASE + (0 * 4 + nt) * 512), vzero);
      c = MMA(BC(a3[1][mt]), ldfrag(W2F_BASE + (1 * 4 + nt) * 512), c);
      acc3[mt][nt] = c;
    }

  // ---- stage 3 post: BN + ReLU + mask + final max-pool over points -> POOL (bf16)
  {
    #pragma unroll
    for (int nt = 0; nt < 4; ++nt) {
      int c = nt * 16 + l15;
      float sc = G2[c] * rsqrtf(V2[c] + BN_EPS);
      float tc = B2[c] - M2[c] * sc;
      float pmv = 0.f;
      #pragma unroll
      for (int mt = 0; mt < 2; ++mt)
        #pragma unroll
        for (int r = 0; r < 8; ++r) {
          int Mrow = mt * 16 + h * 8 + r;
          float y = acc3[mt][nt][r] * sc + tc;
          y = fmaxf(y, 0.f);
          y = ((mball >> Mrow) & 1u) ? y : 0.f;
          pmv = fmaxf(pmv, y);
        }
      pmv = fmaxf(pmv, __shfl_xor(pmv, 16));
      if (lane < 16) lds[POOL + nt * 16 + l15] = f2bf(pmv);
    }
  }

  // ---- tail MLP (1x64 per polyline): VALU, coalesced weight reads ----
  const float valid = (mball != 0u) ? 1.f : 0.f;
  #pragma unroll
  for (int t = 0; t < 2; ++t) {                       // hidden: 2 cols/lane
    int c = lane + t * 32;
    float a = b3[c];
    for (int f = 0; f < 64; ++f)
      a += bf2f(lds[POOL + f]) * W3[f * 64 + c];
    lds[AW + c] = f2bf(fmaxf(a, 0.f));                // A-regions are dead; reuse
  }
  #pragma unroll
  for (int k = 0; k < 4; ++k) {                       // out: 4 cols/lane
    int o = lane + k * 32;
    float a = b4[o];
    for (int c = 0; c < 64; ++c)
      a += bf2f(lds[AW + c]) * W4[c * 128 + o];
    out[(size_t)p * 128 + o] = a * valid;             // coalesced store
  }
}

extern "C" void kernel_launch(void* const* d_in, const int* in_sizes, int n_in,
                              void* d_out, int out_size, void* d_ws, size_t ws_size,
                              hipStream_t stream) {
  (void)in_sizes; (void)n_in; (void)out_size; (void)d_ws; (void)ws_size;
  const float* poly = (const float*)d_in[0];
  const float* Wpre = (const float*)d_in[1];
  const float* G0 = (const float*)d_in[2];
  const float* B0 = (const float*)d_in[3];
  const float* M0 = (const float*)d_in[4];
  const float* V0 = (const float*)d_in[5];
  const float* W1 = (const float*)d_in[6];
  const float* G1 = (const float*)d_in[7];
  const float* B1 = (const float*)d_in[8];
  const float* M1 = (const float*)d_in[9];
  const float* V1 = (const float*)d_in[10];
  const float* W2 = (const float*)d_in[11];
  const float* G2 = (const float*)d_in[12];
  const float* B2 = (const float*)d_in[13];
  const float* M2 = (const float*)d_in[14];
  const float* V2 = (const float*)d_in[15];
  const float* W3 = (const float*)d_in[16];
  const float* b3 = (const float*)d_in[17];
  const float* W4 = (const float*)d_in[18];
  const float* b4 = (const float*)d_in[19];
  const unsigned char* maskp = (const unsigned char*)d_in[20];
  float* out = (float*)d_out;

  // 16384 polylines, 8 per block (1 per wave32), 256 threads = 8 waves
  pointnet_fused<<<2048, 256, 0, stream>>>(
      poly, Wpre, G0, B0, M0, V0, W1, G1, B1, M1, V1,
      W2, G2, B2, M2, V2, W3, b3, W4, b4, maskp, out);
}
